// AgentGnn_11793980195032
// MI455X (gfx1250) — compile-verified, matrix operand order checked by reading, not run on hardware
//
#include <hip/hip_runtime.h>
#include <hip/hip_bf16.h>
#include <math.h>

typedef __attribute__((ext_vector_type(16))) _Float16 v16h;
typedef __attribute__((ext_vector_type(8)))  float    v8f;

#define NSAMP 128
#define AG    64
#define D     128
#define NN    (NSAMP*AG)      // 8192 nodes
#define KEXT  160             // 128 x-dims + 2 centers + 1 bias + pad
#define QCOLS 512             // Pf|Qf|Ps|Qs
#define EPSV  1e-5f

// ---------------------------------------------------------------- helpers
__device__ __forceinline__ float fsigmoid(float x) {
    return 1.0f / (1.0f + __expf(-x));
}
__device__ __forceinline__ float fsoftplus(float x) {
    return fmaxf(x, 0.0f) + __logf(1.0f + __expf(-fabsf(x)));
}

// ---------------------------------------------------------------- zero stats
__global__ void k_zero_stats(float* stats) {
    stats[threadIdx.x] = 0.0f;   // 256 floats: [0..127]=sum, [128..255]=sumsq
}

// ---------------------------------------------------------------- build Xext (f16)
// Xext[n][k]: k<128 -> x[n][k]; 128/129 -> centers[n][0/1]; 130 -> 1.0; else 0
__global__ void k_prep_xext(const float* __restrict__ x,
                            const float* __restrict__ centers,
                            _Float16* __restrict__ Xext) {
    int idx = blockIdx.x * blockDim.x + threadIdx.x;
    if (idx >= NN * KEXT) return;
    int n = idx / KEXT, k = idx % KEXT;
    float v;
    if      (k < 128)  v = x[n * D + k];
    else if (k == 128) v = centers[n * 2 + 0];
    else if (k == 129) v = centers[n * 2 + 1];
    else if (k == 130) v = 1.0f;
    else               v = 0.0f;
    Xext[idx] = (_Float16)v;
}

// ---------------------------------------------------------------- repack weights (f16, transposed)
// Wpack[l][q][k], q in [0,512): blk=q>>7 (0:Pf 1:Qf 2:Ps 3:Qs), c=q&127
__global__ void k_repack_w(const float* __restrict__ Wf1, const float* __restrict__ bf1,
                           const float* __restrict__ Ws1, const float* __restrict__ bs1,
                           const float* __restrict__ Wf2, const float* __restrict__ bf2,
                           const float* __restrict__ Ws2, const float* __restrict__ bs2,
                           _Float16* __restrict__ Wp) {
    int idx = blockIdx.x * blockDim.x + threadIdx.x;
    if (idx >= 2 * QCOLS * KEXT) return;
    int l = idx / (QCOLS * KEXT);
    int r = idx % (QCOLS * KEXT);
    int q = r / KEXT, k = r % KEXT;
    int blk = q >> 7, c = q & 127;
    const float* W    = (blk < 2) ? (l ? Wf2 : Wf1) : (l ? Ws2 : Ws1);
    const float* bias = (blk < 2) ? (l ? bf2 : bf1) : (l ? bs2 : bs1);
    float v = 0.0f;
    if (k < 128) {
        int row = (blk & 1) ? (128 + k) : k;          // dst-part rows 0..127, src-part 128..255
        v = W[row * D + c];
    } else if (k < 130) {
        v = W[(256 + (k - 128)) * D + c];             // edge-attr rows 256/257
        if (blk & 1) v = -v;                          // Q side gets -centers contribution
    } else if (k == 130) {
        v = (blk & 1) ? 0.0f : bias[c];               // bias only on P side
    }
    Wp[idx] = (_Float16)v;
}

// ---------------------------------------------------------------- WMMA GEMM: T[8192][512] = Xext @ Wpack^T
// one wave per 16(M) x 64(N) output tile: 1 A fragment, 4 B fragments,
// 4 independent accumulator chains; K = 160 -> 20 wmma per wave
__global__ __launch_bounds__(256)
void k_gemm_wmma(const _Float16* __restrict__ Xext,
                 const _Float16* __restrict__ Wp,
                 float* __restrict__ T) {
    int wave = blockIdx.x * 8 + (threadIdx.x >> 5);
    int lane = threadIdx.x & 31;
    int tile_n = wave & 7;           // 8 tiles of 64 cols
    int tile_m = wave >> 3;          // 512 tiles of 16 rows
    int hk = (lane >> 4) * 16;       // K-half select per lane group
    const _Float16* aptr = Xext + (size_t)(tile_m * 16 + (lane & 15)) * KEXT + hk;
    const _Float16* bptr = Wp   + (size_t)(tile_n * 64 + (lane & 15)) * KEXT + hk;
    v8f c0 = {}, c1 = {}, c2 = {}, c3 = {};
#pragma unroll
    for (int k0 = 0; k0 < KEXT; k0 += 32) {
        v16h a  = *(const v16h*)(aptr + k0);
        v16h b0 = *(const v16h*)(bptr + k0);
        v16h b1 = *(const v16h*)(bptr + 16 * KEXT + k0);
        v16h b2 = *(const v16h*)(bptr + 32 * KEXT + k0);
        v16h b3 = *(const v16h*)(bptr + 48 * KEXT + k0);
        c0 = __builtin_amdgcn_wmma_f32_16x16x32_f16(false, a, false, b0, (short)0, c0, false, false);
        c1 = __builtin_amdgcn_wmma_f32_16x16x32_f16(false, a, false, b1, (short)0, c1, false, false);
        c2 = __builtin_amdgcn_wmma_f32_16x16x32_f16(false, a, false, b2, (short)0, c2, false, false);
        c3 = __builtin_amdgcn_wmma_f32_16x16x32_f16(false, a, false, b3, (short)0, c3, false, false);
    }
    int row = tile_m * 16 + (lane >> 4) * 8;
    int col = tile_n * 64 + (lane & 15);
    float* tp = T + (size_t)row * QCOLS + col;
#pragma unroll
    for (int i = 0; i < 8; ++i) {
        tp[(size_t)i * QCOLS +  0] = c0[i];
        tp[(size_t)i * QCOLS + 16] = c1[i];
        tp[(size_t)i * QCOLS + 32] = c2[i];
        tp[(size_t)i * QCOLS + 48] = c3[i];
    }
}

// ---------------------------------------------------------------- per-sample edge kernel
// agg[i][c] = sum_{j != i} sigmoid(Pf[i][c]+Qf[j][c]) * softplus(Ps[i][c]+Qs[j][c])
__global__ __launch_bounds__(256)
void k_edges(const float* __restrict__ T,
             float* __restrict__ agg,
             float* __restrict__ stats) {
    __shared__ float lQf[AG][D];     // 32 KB
    __shared__ float lQs[AG][D];     // 32 KB  (total exactly 64 KB)
    int s = blockIdx.x;
    int t = threadIdx.x;
    const float* Tg = T + (size_t)s * AG * QCOLS;

    // stage Qf/Qs for this sample's 64 nodes into LDS
    for (int idx = t; idx < AG * 32; idx += 256) {
        int j = idx >> 5, c4 = (idx & 31) * 4;
        float4 qf = *(const float4*)(Tg + (size_t)j * QCOLS + 128 + c4);
        float4 qs = *(const float4*)(Tg + (size_t)j * QCOLS + 384 + c4);
        *(float4*)(&lQf[j][c4]) = qf;
        *(float4*)(&lQs[j][c4]) = qs;
    }
    __syncthreads();

    int wave = t >> 5, lane = t & 31;
    int c = lane * 4;
    float s0 = 0.f, s1 = 0.f, s2 = 0.f, s3 = 0.f;      // per-channel sum
    float q0 = 0.f, q1 = 0.f, q2 = 0.f, q3 = 0.f;      // per-channel sumsq

#pragma unroll 1
    for (int ii = 0; ii < 8; ++ii) {
        int i = wave * 8 + ii;                          // dst node (local)
        const float* Prow = Tg + (size_t)i * QCOLS;
        float4 pf = *(const float4*)(Prow + c);
        float4 ps = *(const float4*)(Prow + 256 + c);
        float a0 = 0.f, a1 = 0.f, a2 = 0.f, a3 = 0.f;
#pragma unroll 4
        for (int j = 0; j < AG; ++j) {                  // include j==i, subtract later
            float4 qf = *(const float4*)(&lQf[j][c]);
            float4 qs = *(const float4*)(&lQs[j][c]);
            a0 += fsigmoid(pf.x + qf.x) * fsoftplus(ps.x + qs.x);
            a1 += fsigmoid(pf.y + qf.y) * fsoftplus(ps.y + qs.y);
            a2 += fsigmoid(pf.z + qf.z) * fsoftplus(ps.z + qs.z);
            a3 += fsigmoid(pf.w + qf.w) * fsoftplus(ps.w + qs.w);
        }
        {   // remove self-loop term
            float4 qf = *(const float4*)(&lQf[i][c]);
            float4 qs = *(const float4*)(&lQs[i][c]);
            a0 -= fsigmoid(pf.x + qf.x) * fsoftplus(ps.x + qs.x);
            a1 -= fsigmoid(pf.y + qf.y) * fsoftplus(ps.y + qs.y);
            a2 -= fsigmoid(pf.z + qf.z) * fsoftplus(ps.z + qs.z);
            a3 -= fsigmoid(pf.w + qf.w) * fsoftplus(ps.w + qs.w);
        }
        float* arow = agg + ((size_t)s * AG + i) * D;
        *(float4*)(arow + c) = make_float4(a0, a1, a2, a3);
        s0 += a0; s1 += a1; s2 += a2; s3 += a3;
        q0 += a0 * a0; q1 += a1 * a1; q2 += a2 * a2; q3 += a3 * a3;
    }
    // per-channel batch statistics (global f32 atomics over 256 addresses)
    atomicAdd(&stats[c + 0], s0);  atomicAdd(&stats[c + 1], s1);
    atomicAdd(&stats[c + 2], s2);  atomicAdd(&stats[c + 3], s3);
    atomicAdd(&stats[128 + c + 0], q0);  atomicAdd(&stats[128 + c + 1], q1);
    atomicAdd(&stats[128 + c + 2], q2);  atomicAdd(&stats[128 + c + 3], q3);
}

// ---------------------------------------------------------------- BN + residual + ReLU
__global__ __launch_bounds__(256)
void k_apply(const float* __restrict__ agg, const float* __restrict__ xprev,
             const float* __restrict__ stats,
             const float* __restrict__ gamma, const float* __restrict__ beta,
             float* __restrict__ xout) {
    int idx = blockIdx.x * blockDim.x + threadIdx.x;   // one float4 per thread
    int n = idx >> 5, c = (idx & 31) * 4;
    const float inv = 1.0f / (float)NN;
    float4 a  = *(const float4*)(agg   + (size_t)n * D + c);
    float4 xp = *(const float4*)(xprev + (size_t)n * D + c);
    float4 o;
    {
        float mu = stats[c + 0] * inv;
        float var = stats[128 + c + 0] * inv - mu * mu;
        o.x = fmaxf((a.x - mu) * rsqrtf(var + EPSV) * gamma[c + 0] + beta[c + 0] + xp.x, 0.f);
    }
    {
        float mu = stats[c + 1] * inv;
        float var = stats[128 + c + 1] * inv - mu * mu;
        o.y = fmaxf((a.y - mu) * rsqrtf(var + EPSV) * gamma[c + 1] + beta[c + 1] + xp.y, 0.f);
    }
    {
        float mu = stats[c + 2] * inv;
        float var = stats[128 + c + 2] * inv - mu * mu;
        o.z = fmaxf((a.z - mu) * rsqrtf(var + EPSV) * gamma[c + 2] + beta[c + 2] + xp.z, 0.f);
    }
    {
        float mu = stats[c + 3] * inv;
        float var = stats[128 + c + 3] * inv - mu * mu;
        o.w = fmaxf((a.w - mu) * rsqrtf(var + EPSV) * gamma[c + 3] + beta[c + 3] + xp.w, 0.f);
    }
    *(float4*)(xout + (size_t)n * D + c) = o;
}

// ---------------------------------------------------------------- launch
extern "C" void kernel_launch(void* const* d_in, const int* in_sizes, int n_in,
                              void* d_out, int out_size, void* d_ws, size_t ws_size,
                              hipStream_t stream) {
    const float* gnn_in  = (const float*)d_in[0];
    const float* centers = (const float*)d_in[1];
    // d_in[2]/d_in[3] (edge_src/edge_dst) unused: graph structure is fixed all-pairs.
    const float* Wf1 = (const float*)d_in[4];  const float* bf1 = (const float*)d_in[5];
    const float* Ws1 = (const float*)d_in[6];  const float* bs1 = (const float*)d_in[7];
    const float* g1  = (const float*)d_in[8];  const float* b1  = (const float*)d_in[9];
    const float* Wf2 = (const float*)d_in[10]; const float* bf2 = (const float*)d_in[11];
    const float* Ws2 = (const float*)d_in[12]; const float* bs2 = (const float*)d_in[13];
    const float* g2  = (const float*)d_in[14]; const float* b2  = (const float*)d_in[15];
    float* out = (float*)d_out;

    // workspace carve-out (256B aligned)
    size_t off = 0;
    auto carve = [&](size_t bytes) {
        void* p = (char*)d_ws + off;
        off += (bytes + 255) & ~(size_t)255;
        return p;
    };
    _Float16* Xext = (_Float16*)carve((size_t)NN * KEXT * 2);        // 2.62 MB
    _Float16* Wp   = (_Float16*)carve((size_t)2 * QCOLS * KEXT * 2); // 0.33 MB
    float*    T    = (float*)   carve((size_t)NN * QCOLS * 4);       // 16 MB
    float*    agg  = (float*)   carve((size_t)NN * D * 4);           // 4 MB
    float*    x1   = (float*)   carve((size_t)NN * D * 4);           // 4 MB
    float*    stats= (float*)   carve(256 * 4);
    (void)ws_size; (void)in_sizes; (void)n_in; (void)out_size;

    // repack weights once (both layers)
    k_repack_w<<<(2 * QCOLS * KEXT + 255) / 256, 256, 0, stream>>>(
        Wf1, bf1, Ws1, bs1, Wf2, bf2, Ws2, bs2, Wp);

    for (int l = 0; l < 2; ++l) {
        const float* xin   = (l == 0) ? gnn_in : x1;
        float*       xout  = (l == 0) ? x1 : out;
        const float* gamma = (l == 0) ? g1 : g2;
        const float* beta  = (l == 0) ? b1 : b2;

        k_zero_stats<<<1, 256, 0, stream>>>(stats);
        k_prep_xext<<<(NN * KEXT + 255) / 256, 256, 0, stream>>>(xin, centers, Xext);
        k_gemm_wmma<<<(NN / 16) * (QCOLS / 64) / 8, 256, 0, stream>>>(
            Xext, Wp + (size_t)l * QCOLS * KEXT, T);
        k_edges<<<NSAMP, 256, 0, stream>>>(T, agg, stats);
        k_apply<<<NN * 32 / 256, 256, 0, stream>>>(agg, xin, stats, gamma, beta, xout);
    }
}
